// LSC_7516192768935
// MI455X (gfx1250) — matrix operand order repaired and checked by reading, not compiled, not run on hardware
//
#include <hip/hip_runtime.h>

typedef __attribute__((ext_vector_type(4)))  __bf16 v4bf;
typedef __attribute__((ext_vector_type(8)))  __bf16 v8bf;
typedef __attribute__((ext_vector_type(16))) __bf16 v16bf;
typedef __attribute__((ext_vector_type(8)))  float  v8f;

#define B_ROWS 4096
#define DIM    128
#define NCLASS 10000
#define NPROXY 3
#define KROWS  (NCLASS * NPROXY)   // 30000

// ---------------------------------------------------------------------------
// Kernel 1: L2-normalize rows of a (nrows x 128) f32 matrix, emit bf16.
// One wave32 per row; each lane handles 4 consecutive floats (32*4 = 128).
// ---------------------------------------------------------------------------
__global__ __launch_bounds__(256) void lsc_normalize_bf16(
    const float* __restrict__ in, unsigned short* __restrict__ out, int nrows)
{
    int wave = (blockIdx.x * blockDim.x + threadIdx.x) >> 5;
    int lane = threadIdx.x & 31;
    if (wave >= nrows) return;

    const float4* row = (const float4*)(in + (size_t)wave * DIM);
    float4 v = row[lane];
    float ss = v.x * v.x + v.y * v.y + v.z * v.z + v.w * v.w;
    // wave32 butterfly reduction
    #pragma unroll
    for (int m = 16; m >= 1; m >>= 1) ss += __shfl_xor(ss, m, 32);

    float inv = 1.0f / fmaxf(sqrtf(ss), 1e-8f);

    v4bf r;
    r.x = (__bf16)(v.x * inv);
    r.y = (__bf16)(v.y * inv);
    r.z = (__bf16)(v.z * inv);
    r.w = (__bf16)(v.w * inv);
    *(v4bf*)((__bf16*)out + (size_t)wave * DIM + lane * 4) = r;
}

// ---------------------------------------------------------------------------
// Kernel 2: fused cosine-GEMM + softmax-over-proxies.
//   sims(32 x 48) per block = xn_tile(32 x 128) * wn_tile(48 x 128)^T
//   6 waves in a 2(M) x 3(N) arrangement; each wave: one 16x16 f32
//   accumulator fed by 4x v_wmma_f32_16x16x32_bf16 (K = 128).
// Fragments are built straight from global memory per the CDNA5 wave32
// VGPR layouts (ISA 7.12.2):
//   A (16x32 bf16): lane m = l%16, half = l/16 holds K chunks
//                   [kb + 8*half, +8) in v0..3 and [kb + 16 + 8*half, +8) in v4..7
//   B (32x16 bf16): lane n = l%16, half = l/16 holds K chunks
//                   [kb + 8*half, +8) in v0..3 and [kb + 16 + 8*half, +8) in v4..7 (same as A)
// ---------------------------------------------------------------------------
#define SIM_LD 52   // padded LDS row stride (floats): 8*52 % 64 != 0 -> no alias

__global__ __launch_bounds__(192) void lsc_gemm_softmax(
    const unsigned short* __restrict__ xn_bf16,   // 4096 x 128
    const unsigned short* __restrict__ wn_bf16,   // 30000 x 128
    float* __restrict__ out)                      // 4096 x 10000
{
    __shared__ float sims[32][SIM_LD];

    const int wv   = threadIdx.x >> 5;   // 0..5
    const int lane = threadIdx.x & 31;
    const int m    = lane & 15;
    const int half = lane >> 4;

    const int mwave = wv / 3;            // 0..1  (M sub-tile)
    const int nwave = wv % 3;            // 0..2  (N sub-tile)

    const int rowM  = blockIdx.y * 32 + mwave * 16;      // xn row base
    const int rowWn = blockIdx.x * 48 + nwave * 16;      // wn row base

    const __bf16* arow = (const __bf16*)xn_bf16 + (size_t)(rowM + m)  * DIM;
    const __bf16* brow = (const __bf16*)wn_bf16 + (size_t)(rowWn + m) * DIM;

    v8f acc = {};

    #pragma unroll
    for (int ks = 0; ks < 4; ++ks) {
        const int kb = ks * 32;

        // A fragment: two 16-byte chunks of row (rowM + m)
        v8bf a0 = *(const v8bf*)(arow + kb + 8 * half);
        v8bf a1 = *(const v8bf*)(arow + kb + 16 + 8 * half);
        v16bf A = __builtin_shufflevector(a0, a1,
                    0, 1, 2, 3, 4, 5, 6, 7, 8, 9, 10, 11, 12, 13, 14, 15);

        // B fragment: two 16-byte chunks of wn row (rowWn + m), same K order as A
        v8bf b0 = *(const v8bf*)(brow + kb + 8 * half);
        v8bf b1 = *(const v8bf*)(brow + kb + 16 + 8 * half);
        v16bf Bf = __builtin_shufflevector(b0, b1,
                    0, 1, 2, 3, 4, 5, 6, 7, 8, 9, 10, 11, 12, 13, 14, 15);

        acc = __builtin_amdgcn_wmma_f32_16x16x32_bf16(
                  false, A, false, Bf, (short)0, acc, false, false);
    }

    // C/D layout: VGPR r, lane l -> row = r + 8*(l/16), col = l%16
    {
        const int mbase = mwave * 16 + 8 * half;
        const int nbase = nwave * 16 + m;
        #pragma unroll
        for (int r = 0; r < 8; ++r)
            sims[mbase + r][nbase] = acc[r];
    }

    __syncthreads();

    // Softmax over the 3 proxies of each class, weighted sum, stream out.
    const int rowsBase  = blockIdx.y * 32;
    const int classBase = blockIdx.x * 16;
    for (int idx = threadIdx.x; idx < 32 * 16; idx += 192) {
        const int r = idx >> 4;
        const int c = idx & 15;
        float s0 = sims[r][3 * c + 0];
        float s1 = sims[r][3 * c + 1];
        float s2 = sims[r][3 * c + 2];
        float mx = fmaxf(s0, fmaxf(s1, s2));
        float e0 = __expf(s0 - mx);
        float e1 = __expf(s1 - mx);
        float e2 = __expf(s2 - mx);
        float res = (e0 * s0 + e1 * s1 + e2 * s2) / (e0 + e1 + e2);
        __builtin_nontemporal_store(
            res, out + (size_t)(rowsBase + r) * NCLASS + classBase + c);
    }
}

// ---------------------------------------------------------------------------
extern "C" void kernel_launch(void* const* d_in, const int* in_sizes, int n_in,
                              void* d_out, int out_size, void* d_ws, size_t ws_size,
                              hipStream_t stream)
{
    const float* x = (const float*)d_in[0];   // (4096, 128)
    const float* w = (const float*)d_in[1];   // (10000, 384) == (30000, 128)
    float* out = (float*)d_out;               // (4096, 10000)

    unsigned short* xn = (unsigned short*)d_ws;                       // 1.0 MB
    unsigned short* wn = xn + (size_t)B_ROWS * DIM;                   // 7.7 MB

    // 8 rows per 256-thread block (one wave32 per row)
    lsc_normalize_bf16<<<B_ROWS / 8, 256, 0, stream>>>(x, xn, B_ROWS);
    lsc_normalize_bf16<<<KROWS / 8, 256, 0, stream>>>(w, wn, KROWS);

    // 625 x 128 blocks, each: 32 rows x 16 classes (48 wn rows)
    dim3 grid(NCLASS / 16, B_ROWS / 32);
    lsc_gemm_softmax<<<grid, 192, 0, stream>>>(xn, wn, out);
}
